// MixHop_lr_84954453115008
// MI455X (gfx1250) — matrix operand order repaired and checked by reading, not compiled, not run on hardware
//
#include <hip/hip_runtime.h>
#include <math.h>

#define IN_F   128
#define HID_F  64
#define OUT_F  40
#define NPOW   3
#define CAT_F  (HID_F * NPOW)   // 192
#define LN_EPS 1e-5f

typedef __attribute__((ext_vector_type(16))) __bf16       v16bf;
typedef __attribute__((ext_vector_type(8)))  float        v8f;
typedef __attribute__((ext_vector_type(8)))  unsigned int v8u;

// Native cast -> hardware packed bf16 convert (RNE) instead of manual bit math.
static __device__ __forceinline__ __bf16 f2bf(float f) { return (__bf16)f; }

static __device__ __forceinline__ float gelu_exact(float x) {
  return 0.5f * x * (1.0f + erff(x * 0.70710678118654752f));
}

// ---------------------------------------------------------------------------
// degree / normalization
// ---------------------------------------------------------------------------
__global__ void k_zero_f(float* p, int n) {
  int t = blockIdx.x * blockDim.x + threadIdx.x;
  if (t < n) p[t] = 0.0f;
}

__global__ void k_deg_acc(const int* __restrict__ dst, float* __restrict__ cnt, int E) {
  int t = blockIdx.x * blockDim.x + threadIdx.x;
  if (t < E) atomicAdd(cnt + dst[t], 1.0f);
}

__global__ void k_make_dis(float* d, int N) {  // deg includes +1 self loop -> always > 0
  int t = blockIdx.x * blockDim.x + threadIdx.x;
  if (t < N) d[t] = rsqrtf(d[t] + 1.0f);
}

// ---------------------------------------------------------------------------
// pack weights (f32 row-major [K][64]) into bf16 WMMA B-fragments:
// frag index = ((kstep*4 + ntile)*32 + lane)*16 + i
// B element (lane, i) <- W[k][n], n = ntile*16 + (lane&15),
//                        k = kstep*32 + ((lane>>4)<<3) + i + (i&8)
// ---------------------------------------------------------------------------
__global__ void k_pack_weights(const float* __restrict__ W1, const float* __restrict__ Wc,
                               __bf16* __restrict__ W1p, __bf16* __restrict__ Wcp) {
  const int W1_ELEMS = 4 * 4 * 32 * 16;  // K=128 -> 8192
  const int WC_ELEMS = 2 * 4 * 32 * 16;  // K=64  -> 4096
  int idx = blockIdx.x * blockDim.x + threadIdx.x;
  if (idx < W1_ELEMS) {
    int rem = idx;
    int i = rem & 15, lane = (rem >> 4) & 31, ntile = (rem >> 9) & 3, kstep = rem >> 11;
    int n = ntile * 16 + (lane & 15);
    int k = kstep * 32 + ((lane >> 4) << 3) + i + (i & 8);
    W1p[idx] = f2bf(W1[k * HID_F + n]);
  } else if (idx < W1_ELEMS + NPOW * WC_ELEMS) {
    int t = idx - W1_ELEMS;
    int p = t / WC_ELEMS;            // 0,1,2 -> powers 6,8,10
    int rem = t % WC_ELEMS;
    int i = rem & 15, lane = (rem >> 4) & 31, ntile = (rem >> 9) & 3, kstep = rem >> 11;
    int n = ntile * 16 + (lane & 15);
    int k = kstep * 32 + ((lane >> 4) << 3) + i + (i & 8);
    int pw = 6 + 2 * p;
    Wcp[t] = f2bf(Wc[(pw * HID_F + k) * HID_F + n]);
  }
}

// A-fragment: 16x32 bf16 tile of f32 matrix with row stride S.
// lane row = rowbase + (lane&15); k = kb + i + (i&8), kb = kstep*32 + (lane>>4)*8
static __device__ __forceinline__ v16bf load_a_frag(const float* __restrict__ A,
                                                    int row, int S, int kstep, int lane) {
  const float* ap = A + (size_t)row * S + kstep * 32 + ((lane >> 4) << 3);
  float4 q0 = *(const float4*)(ap + 0);
  float4 q1 = *(const float4*)(ap + 4);
  float4 q2 = *(const float4*)(ap + 16);
  float4 q3 = *(const float4*)(ap + 20);
  v16bf a;
  a[0]  = f2bf(q0.x); a[1]  = f2bf(q0.y); a[2]  = f2bf(q0.z); a[3]  = f2bf(q0.w);
  a[4]  = f2bf(q1.x); a[5]  = f2bf(q1.y); a[6]  = f2bf(q1.z); a[7]  = f2bf(q1.w);
  a[8]  = f2bf(q2.x); a[9]  = f2bf(q2.y); a[10] = f2bf(q2.z); a[11] = f2bf(q2.w);
  a[12] = f2bf(q3.x); a[13] = f2bf(q3.y); a[14] = f2bf(q3.z); a[15] = f2bf(q3.w);
  return a;
}

// ---------------------------------------------------------------------------
// GEMM1: h0 = LayerNorm(gelu(x @ W1 + b1)); one wave per 16-row tile, N=64, K=128
// ---------------------------------------------------------------------------
__global__ void k_gemm1_gelu_ln(const float* __restrict__ x, const __bf16* __restrict__ W1p,
                                const float* __restrict__ b1, const float* __restrict__ g1,
                                const float* __restrict__ be1, float* __restrict__ h0, int ntilesM) {
  int wave = (blockIdx.x * blockDim.x + threadIdx.x) >> 5;
  int lane = threadIdx.x & 31;
  if (wave >= ntilesM) return;            // wave-uniform
  int rowbase = wave * 16;
  int arow = rowbase + (lane & 15);

  v16bf A[4];
#pragma unroll
  for (int ks = 0; ks < 4; ++ks) A[ks] = load_a_frag(x, arow, IN_F, ks, lane);

  v8f acc[4];
#pragma unroll
  for (int nt = 0; nt < 4; ++nt) {
    int col = nt * 16 + (lane & 15);
    float bias = b1[col];
    v8f c;
#pragma unroll
    for (int j = 0; j < 8; ++j) c[j] = bias;
#pragma unroll
    for (int ks = 0; ks < 4; ++ks) {
      const v8u* bp = (const v8u*)(W1p + (((ks * 4 + nt) * 32 + lane) << 4));
      v16bf B = __builtin_bit_cast(v16bf, *bp);
      c = __builtin_amdgcn_wmma_f32_16x16x32_bf16(false, A[ks], false, B,
                                                  (short)0, c, false, false);
    }
#pragma unroll
    for (int j = 0; j < 8; ++j) c[j] = gelu_exact(c[j]);
    acc[nt] = c;
  }

  float gl[4], bl[4];
#pragma unroll
  for (int nt = 0; nt < 4; ++nt) {
    int col = nt * 16 + (lane & 15);
    gl[nt] = g1[col];
    bl[nt] = be1[col];
  }

  // LayerNorm over 64 features of each row: reduce across N-tiles (regs) then 16 lanes
#pragma unroll
  for (int j = 0; j < 8; ++j) {
    float sm = 0.0f, sq = 0.0f;
#pragma unroll
    for (int nt = 0; nt < 4; ++nt) { float v = acc[nt][j]; sm += v; sq += v * v; }
#pragma unroll
    for (int m = 1; m < 16; m <<= 1) { sm += __shfl_xor(sm, m); sq += __shfl_xor(sq, m); }
    float mean = sm * (1.0f / 64.0f);
    float var  = sq * (1.0f / 64.0f) - mean * mean;
    float rstd = rsqrtf(var + LN_EPS);
    int r = rowbase + j + ((lane >> 4) << 3);
#pragma unroll
    for (int nt = 0; nt < 4; ++nt) {
      float v = (acc[nt][j] - mean) * rstd * gl[nt] + bl[nt];
      h0[(size_t)r * HID_F + nt * 16 + (lane & 15)] = v;
    }
  }
}

// ---------------------------------------------------------------------------
// GEMM2: hcat[:, colofs:colofs+64] = cur @ Wc[p] + bc[p];  K=64, N=64
// ---------------------------------------------------------------------------
__global__ void k_gemm2(const float* __restrict__ cur, const __bf16* __restrict__ Wp,
                        const float* __restrict__ bias, float* __restrict__ hcat,
                        int colofs, int ntilesM) {
  int wave = (blockIdx.x * blockDim.x + threadIdx.x) >> 5;
  int lane = threadIdx.x & 31;
  if (wave >= ntilesM) return;
  int rowbase = wave * 16;
  int arow = rowbase + (lane & 15);

  v16bf A[2];
#pragma unroll
  for (int ks = 0; ks < 2; ++ks) A[ks] = load_a_frag(cur, arow, HID_F, ks, lane);

#pragma unroll
  for (int nt = 0; nt < 4; ++nt) {
    int col = nt * 16 + (lane & 15);
    float b = bias[col];
    v8f c;
#pragma unroll
    for (int j = 0; j < 8; ++j) c[j] = b;
#pragma unroll
    for (int ks = 0; ks < 2; ++ks) {
      const v8u* bp = (const v8u*)(Wp + (((ks * 4 + nt) * 32 + lane) << 4));
      v16bf B = __builtin_bit_cast(v16bf, *bp);
      c = __builtin_amdgcn_wmma_f32_16x16x32_bf16(false, A[ks], false, B,
                                                  (short)0, c, false, false);
    }
#pragma unroll
    for (int j = 0; j < 8; ++j) {
      int r = rowbase + j + ((lane >> 4) << 3);
      hcat[(size_t)r * CAT_F + colofs + col] = c[j];
    }
  }
}

// ---------------------------------------------------------------------------
// propagate: out[i] = dis[i]^2 * in[i] + sum_{e: dst==i} dis[src]*dis[dst]*in[src]
// ---------------------------------------------------------------------------
__global__ void k_prop_init(const float* __restrict__ hin, const float* __restrict__ dis,
                            float* __restrict__ hout, int N) {
  int tid = blockIdx.x * blockDim.x + threadIdx.x;   // N*16 threads, float4 each
  if (tid >= N * 16) return;
  int row = tid >> 4, c4 = tid & 15;
  float d = dis[row];
  float w = d * d;
  float4 v = ((const float4*)(hin + (size_t)row * HID_F))[c4];
  float4 o = make_float4(w * v.x, w * v.y, w * v.z, w * v.w);
  ((float4*)(hout + (size_t)row * HID_F))[c4] = o;
}

__global__ void k_prop_edges(const int* __restrict__ src, const int* __restrict__ dst,
                             const float* __restrict__ dis, const float* __restrict__ hin,
                             float* __restrict__ hout, int E) {
  int tid = blockIdx.x * blockDim.x + threadIdx.x;   // 16 lanes per edge
  int e = tid >> 4, c4 = tid & 15;
  if (e >= E) return;
  int lane = threadIdx.x & 31;
  // Only the group leader (lane 0 / lane 16) loads edge scalars; broadcast in-wave.
  int s = 0, d = 0;
  float w = 0.0f;
  if ((lane & 15) == 0) {
    s = src[e];
    d = dst[e];
    w = dis[s] * dis[d];
  }
  int bsrc = lane & 16;        // 0 for lanes 0-15, 16 for lanes 16-31
  s = __shfl(s, bsrc);
  d = __shfl(d, bsrc);
  w = __shfl(w, bsrc);
  float4 v = ((const float4*)(hin + (size_t)s * HID_F))[c4];
  float* o = hout + (size_t)d * HID_F + c4 * 4;
  atomicAdd(o + 0, w * v.x);
  atomicAdd(o + 1, w * v.y);
  atomicAdd(o + 2, w * v.z);
  atomicAdd(o + 3, w * v.w);
}

// ---------------------------------------------------------------------------
// final: out = LayerNorm(gelu(hcat)) @ W2 + b2   (192 -> 40)
// ---------------------------------------------------------------------------
#define K6_WAVES 8
__global__ void k_final_ln_gemm(const float* __restrict__ hcat, const float* __restrict__ g2,
                                const float* __restrict__ be2, const float* __restrict__ W2,
                                const float* __restrict__ b2, float* __restrict__ out, int N) {
  __shared__ float sW[CAT_F * OUT_F];  // 30 KB
  __shared__ float sB[OUT_F];
  for (int i = threadIdx.x; i < CAT_F * OUT_F; i += blockDim.x) sW[i] = W2[i];
  for (int i = threadIdx.x; i < OUT_F; i += blockDim.x) sB[i] = b2[i];
  __syncthreads();

  int w = threadIdx.x >> 5, lane = threadIdx.x & 31;
  for (int row = blockIdx.x * K6_WAVES + w; row < N; row += gridDim.x * K6_WAVES) {
    float v[6];
    float sm = 0.0f, sq = 0.0f;
#pragma unroll
    for (int k = 0; k < 6; ++k) {
      float t = hcat[(size_t)row * CAT_F + k * 32 + lane];
      t = gelu_exact(t);
      v[k] = t; sm += t; sq += t * t;
    }
#pragma unroll
    for (int m = 1; m < 32; m <<= 1) { sm += __shfl_xor(sm, m); sq += __shfl_xor(sq, m); }
    float mean = sm * (1.0f / 192.0f);
    float var  = sq * (1.0f / 192.0f) - mean * mean;
    float rstd = rsqrtf(var + LN_EPS);
#pragma unroll
    for (int k = 0; k < 6; ++k) {
      int f = k * 32 + lane;
      v[k] = (v[k] - mean) * rstd * g2[f] + be2[f];
    }
    int c1 = 32 + (lane & 7);
    float a0 = sB[lane];
    float a1 = sB[c1];
#pragma unroll
    for (int k = 0; k < 6; ++k) {
#pragma unroll
      for (int l = 0; l < 32; ++l) {
        float hk = __shfl(v[k], l);
        int kk = k * 32 + l;
        a0 += hk * sW[kk * OUT_F + lane];
        a1 += hk * sW[kk * OUT_F + c1];
      }
    }
    out[(size_t)row * OUT_F + lane] = a0;
    if (lane < 8) out[(size_t)row * OUT_F + c1] = a1;
  }
}

// ---------------------------------------------------------------------------
extern "C" void kernel_launch(void* const* d_in, const int* in_sizes, int n_in,
                              void* d_out, int out_size, void* d_ws, size_t ws_size,
                              hipStream_t stream) {
  const float* x   = (const float*)d_in[0];
  const int*   ei  = (const int*)d_in[1];
  const float* W1  = (const float*)d_in[2];
  const float* b1  = (const float*)d_in[3];
  const float* Wc  = (const float*)d_in[4];
  const float* bc  = (const float*)d_in[5];
  const float* W2  = (const float*)d_in[6];
  const float* b2  = (const float*)d_in[7];
  const float* g1  = (const float*)d_in[8];
  const float* be1 = (const float*)d_in[9];
  const float* g2  = (const float*)d_in[10];
  const float* be2 = (const float*)d_in[11];

  const int N = in_sizes[0] / IN_F;
  const int E = in_sizes[1] / 2;
  const int* src = ei;
  const int* dst = ei + E;

  char* ws = (char*)d_ws;
  size_t off = 0;
  auto carve = [&](size_t bytes) -> char* {
    char* p = ws + off;
    off = (off + bytes + 255) & ~(size_t)255;
    return p;
  };
  float*  dis  = (float*)carve((size_t)N * 4);
  float*  bufA = (float*)carve((size_t)N * HID_F * 4);
  float*  bufB = (float*)carve((size_t)N * HID_F * 4);
  float*  hcat = (float*)carve((size_t)N * CAT_F * 4);
  __bf16* W1p  = (__bf16*)carve(8192 * 2);
  __bf16* Wcp  = (__bf16*)carve((size_t)NPOW * 4096 * 2);

  // 1) symmetric GCN normalization
  k_zero_f<<<(N + 255) / 256, 256, 0, stream>>>(dis, N);
  k_deg_acc<<<(E + 255) / 256, 256, 0, stream>>>(dst, dis, E);
  k_make_dis<<<(N + 255) / 256, 256, 0, stream>>>(dis, N);

  // 2) weights -> bf16 WMMA fragments
  int packElems = 8192 + NPOW * 4096;
  k_pack_weights<<<(packElems + 255) / 256, 256, 0, stream>>>(W1, Wc, W1p, Wcp);

  // 3) h0 = LN(gelu(x @ W1 + b1))
  int tilesM = N / 16;   // N = 100000 -> 6250 exact
  k_gemm1_gelu_ln<<<(tilesM + 1) / 2, 64, 0, stream>>>(x, W1p, b1, g1, be1, bufA, tilesM);

  // 4) 10 hops of propagation; powers 6/8/10 land in bufA (even hop count)
  float* cur = bufA;
  float* nxt = bufB;
  for (int j = 1; j <= 10; ++j) {
    k_prop_init<<<(N * 16 + 255) / 256, 256, 0, stream>>>(cur, dis, nxt, N);
    k_prop_edges<<<(E * 16 + 255) / 256, 256, 0, stream>>>(src, dst, dis, cur, nxt, E);
    float* t = cur; cur = nxt; nxt = t;
    if (j == 6 || j == 8 || j == 10) {
      int pi = (j - 6) / 2;
      k_gemm2<<<(tilesM + 1) / 2, 64, 0, stream>>>(cur, Wcp + (size_t)pi * 4096,
                                                   bc + (size_t)j * HID_F, hcat,
                                                   pi * HID_F, tilesM);
    }
  }

  // 5) out = LN(gelu(hcat)) @ W2 + b2
  k_final_ln_gemm<<<512, 256, 0, stream>>>(hcat, g2, be2, W2, b2, (float*)d_out, N);
}
